// FocalCTCLoss_11424613007497
// MI455X (gfx1250) — compile-verified, matrix operand order checked by reading, not run on hardware
//
#include <hip/hip_runtime.h>
#include <hip/hip_bf16.h>

#define EPS      1e-7f
#define NEG_INF  (-1e30f)
#define LN2      0.69314718055994530942f
#define B_DIM    256
#define T_DIM    1024
#define V_DIM    128
#define L_DIM    64
#define BLANK    127

// Base-2 exponential: v_exp_f32 is natively 2^x on AMDGPU.
#if __has_builtin(__builtin_amdgcn_exp2f)
#define EXP2F(x) __builtin_amdgcn_exp2f(x)
#else
#define EXP2F(x) exp2f(x)
#endif

typedef __attribute__((ext_vector_type(2)))  float    v2f;
typedef __attribute__((ext_vector_type(8)))  float    v8f;
typedef __attribute__((ext_vector_type(16))) _Float16 v16h;

// ---------------------------------------------------------------------------
// Kernel 1: lognorm2[b*T+t] = log2( sum_v y_pred[b,t,v] + V*EPS )   (log2 domain)
// Row-sum over V=128 on the WMMA pipe (ones-matrix multiply, f32 exact),
// two interleaved accumulators to halve the D->C dependency chain.
// Bandwidth-bound: 134 MB read, 1 MB written (~6 us at 23.3 TB/s).
// 128 threads (4 waves) per block, 64 rows per block, 16 rows per wave.
// ---------------------------------------------------------------------------
__global__ __launch_bounds__(128)
void k_rowsum_lognorm(const float* __restrict__ yp, float* __restrict__ lognorm2) {
    const int ROWS = 64, STRIDE = 132; // 528 B row stride: bank-conflict-free A reads
    __shared__ __align__(16) float lds[ROWS * STRIDE];

    const int tid        = threadIdx.x;
    const int block_row0 = blockIdx.x * ROWS;

    // Cooperative, fully-coalesced fill: 64 rows * 32 float4 = 2048 float4.
    const float4* src = (const float4*)(yp + (size_t)block_row0 * V_DIM);
    #pragma unroll
    for (int it = 0; it < 16; ++it) {
        int idx = it * 128 + tid;
        int r   = idx >> 5;
        int c4  = idx & 31;
        float4 v = src[r * 32 + c4];
        *(float4*)(&lds[r * STRIDE + c4 * 4]) = v; // 528r + 16c : 16B aligned
    }
    __syncthreads();

    const int wave  = tid >> 5;
    const int lane  = tid & 31;
    const int rbase = wave * 16;            // 16 rows handled by this wave
    const int lrow  = lane & 15;            // A-matrix M = lane % 16

    v8f c0 = {0.f, 0.f, 0.f, 0.f, 0.f, 0.f, 0.f, 0.f};
    v8f c1 = {0.f, 0.f, 0.f, 0.f, 0.f, 0.f, 0.f, 0.f};

#if __has_builtin(__builtin_amdgcn_wmma_f32_16x16x4_f32)
    // A: 16x4 f32. lanes 0-15 supply K=0,1; lanes 16-31 supply K=2,3.
    const int koff = (lane < 16) ? 0 : 2;
    const float* myrow = &lds[(rbase + lrow) * STRIDE + koff];
    v2f bones; bones.x = 1.0f; bones.y = 1.0f;   // B = ones: layout-agnostic
    #pragma unroll
    for (int kk = 0; kk < 32; kk += 2) {         // 32 * K4 = 128 = V
        v2f a0, a1;
        a0.x = myrow[4 * kk + 0];
        a0.y = myrow[4 * kk + 1];
        a1.x = myrow[4 * kk + 4];
        a1.y = myrow[4 * kk + 5];
        c0 = __builtin_amdgcn_wmma_f32_16x16x4_f32(
                 false, a0, false, bones, (short)0, c0, false, false);
        c1 = __builtin_amdgcn_wmma_f32_16x16x4_f32(
                 false, a1, false, bones, (short)0, c1, false, false);
    }
#else
    // Fallback: codegen-confirmed f16 WMMA (f32 accumulate).
    const int koff = (lane < 16) ? 0 : 8;
    const float* myrow = &lds[(rbase + lrow) * STRIDE];
    v16h bones;
    #pragma unroll
    for (int j = 0; j < 16; ++j) bones[j] = (_Float16)1.0f;
    #pragma unroll
    for (int kk = 0; kk < 4; kk += 2) {          // 4 * K32 = 128 = V
        v16h a0, a1;
        #pragma unroll
        for (int j = 0; j < 8; ++j) {
            a0[j]     = (_Float16)myrow[32 * kk + koff + j];
            a0[8 + j] = (_Float16)myrow[32 * kk + 16 + koff + j];
            a1[j]     = (_Float16)myrow[32 * (kk + 1) + koff + j];
            a1[8 + j] = (_Float16)myrow[32 * (kk + 1) + 16 + koff + j];
        }
        c0 = __builtin_amdgcn_wmma_f32_16x16x32_f16(
                 false, a0, false, bones, (short)0, c0, false, false);
        c1 = __builtin_amdgcn_wmma_f32_16x16x32_f16(
                 false, a1, false, bones, (short)0, c1, false, false);
    }
#endif

    // D column 0: lane 0 holds rows rbase+0..7; lane 16 holds rows rbase+8..15.
    if ((lane & 15) == 0) {
        const int roff = (lane == 0) ? 0 : 8;
        float* dst = &lognorm2[block_row0 + rbase + roff];
        #pragma unroll
        for (int i = 0; i < 8; ++i)
            dst[i] = __log2f((c0[i] + c1[i]) + (float)V_DIM * EPS);
    }
}

// ---------------------------------------------------------------------------
// Kernel 2: CTC forward DP + focal transform, one wave32 per batch element.
// 129 states register-resident: 5 states/lane over lanes 0..25.
// Cross-lane s-1 / s-2 via __shfl_up -> zero barriers, zero LDS.
// Entire DP in log2 domain: v_exp_f32 / v_log_f32 used natively, no ln2 muls
// on the 1024-step serial chain. Gathers software-pipelined one step ahead.
// ---------------------------------------------------------------------------
__global__ __launch_bounds__(32)
void k_ctc_dp(const int* __restrict__ y_true, const float* __restrict__ yp,
              const float* __restrict__ lognorm2, float* __restrict__ focal_out) {
    const int b    = blockIdx.x;
    const int lane = threadIdx.x;            // wave32
    const int* lab = y_true + b * L_DIM;

    // Per-lane extended-label setup: states s = 5*lane + j
    int  el[5];
    bool ok[5];
    #pragma unroll
    for (int j = 0; j < 5; ++j) {
        const int s = 5 * lane + j;
        int  e   = BLANK;
        bool osk = false;
        if (s < 129 && (s & 1)) {            // odd state = label
            e = lab[(s - 1) >> 1];
            const int em2 = (s >= 3) ? lab[(s - 3) >> 1] : BLANK;
            osk = (e != BLANK) && (e != em2);
        }
        el[j] = e;
        ok[j] = osk;
    }

    const float* row = yp + (size_t)b * T_DIM * V_DIM;
    const float* ln  = lognorm2 + b * T_DIM;

    // t = 0: alpha2[0..1] = lp2, rest NEG_INF   (alpha kept in log2 units)
    float st[5];
    {
        const float ln0 = ln[0];
        #pragma unroll
        for (int j = 0; j < 5; ++j) {
            const int s   = 5 * lane + j;
            const float lp = __log2f(row[el[j]] + EPS) - ln0;
            st[j] = (s < 2) ? lp : NEG_INF;
        }
    }

    // Prime the pipeline with t = 1 gathers.
    float rawc[5], lnc;
    {
        const float* r1 = row + 1 * V_DIM;
        #pragma unroll
        for (int j = 0; j < 5; ++j) rawc[j] = r1[el[j]];
        lnc = ln[1];
    }

    for (int t = 1; t < T_DIM; ++t) {
        // Issue next step's gathers now; they sit off the serial DP chain.
        const int tn = (t + 1 < T_DIM) ? (t + 1) : t;
        const float* rn = row + tn * V_DIM;
        float rawn[5];
        #pragma unroll
        for (int j = 0; j < 5; ++j) rawn[j] = rn[el[j]];
        const float lnn = ln[tn];

        // Neighbor states from previous lane: s-1 = prev st[4], s-2 = prev st[3].
        float pm1 = __shfl_up(st[4], 1, 32);
        float pm2 = __shfl_up(st[3], 1, 32);
        if (lane == 0) { pm1 = NEG_INF; pm2 = NEG_INF; }

        float ns[5];
        #pragma unroll
        for (int j = 0; j < 5; ++j) {
            const float a0 = st[j];
            const float a1 = (j == 0) ? pm1 : st[j - 1];
            float       a2 = (j == 0) ? pm2 : ((j == 1) ? pm1 : st[j - 2]);
            a2 = ok[j] ? a2 : NEG_INF;
            const float m   = fmaxf(a0, fmaxf(a1, a2));
            const float sum = EXP2F(a0 - m) + EXP2F(a1 - m) + EXP2F(a2 - m);
            const float lp  = __log2f(rawc[j] + EPS) - lnc;
            ns[j] = m + __log2f(sum) + lp;
        }
        #pragma unroll
        for (int j = 0; j < 5; ++j) { st[j] = ns[j]; rawc[j] = rawn[j]; }
        lnc = lnn;
    }

    // alphaT[S-1] = state 128 = lane25.st[3]; alphaT[S-2] = state 127 = lane25.st[2]
    if (lane == 25) {
        const float aA  = st[3], aB = st[2];
        const float m   = fmaxf(aA, aB);
        const float ll2 = m + __log2f(EXP2F(aA - m) + EXP2F(aB - m)); // log2 lik
        const float loss = -ll2 * LN2;          // natural-log CTC loss
        const float p    = EXP2F(ll2);          // exp(-loss) = 2^{ll2}
        const float om   = 1.0f - p;
        focal_out[b] = 0.25f * om * om * loss;
    }
}

// ---------------------------------------------------------------------------
// Kernel 3: mean over B=256 focal values -> scalar output
// ---------------------------------------------------------------------------
__global__ __launch_bounds__(256)
void k_reduce_mean(const float* __restrict__ focal, float* __restrict__ out) {
    __shared__ float s[256];
    const int tid = threadIdx.x;
    s[tid] = focal[tid];
    __syncthreads();
    for (int w = 128; w > 0; w >>= 1) {
        if (tid < w) s[tid] += s[tid + w];
        __syncthreads();
    }
    if (tid == 0) out[0] = s[0] * (1.0f / (float)B_DIM);
}

// ---------------------------------------------------------------------------
extern "C" void kernel_launch(void* const* d_in, const int* in_sizes, int n_in,
                              void* d_out, int out_size, void* d_ws, size_t ws_size,
                              hipStream_t stream) {
    const int*   y_true = (const int*)d_in[0];    // [B, L] int32
    const float* y_pred = (const float*)d_in[1];  // [B, T, V] float32

    float* lognorm2 = (float*)d_ws;                      // B*T floats = 1 MB
    float* focal    = lognorm2 + (size_t)B_DIM * T_DIM;  // B floats

    const int rows_total = B_DIM * T_DIM;                // 262144
    k_rowsum_lognorm<<<rows_total / 64, 128, 0, stream>>>(y_pred, lognorm2);
    k_ctc_dp<<<B_DIM, 32, 0, stream>>>(y_true, y_pred, lognorm2, focal);
    k_reduce_mean<<<1, 256, 0, stream>>>(focal, (float*)d_out);
}
